// CitationClassifier_12000138625436
// MI455X (gfx1250) — matrix operand order repaired and checked by reading, not compiled, not run on hardware
//
#include <hip/hip_runtime.h>
#include <hip/hip_bf16.h>
#include <float.h>

// ---------------------------------------------------------------------------
// CDNA5 (gfx1250) citation-classifier forward pass.
// bf16 WMMA (v_wmma_f32_16x16x32_bf16) for all big GEMMs, f32 accumulate.
// Weights pre-converted once per launch to padded transposed bf16.
// GEMM staging uses gfx1250 async global->LDS copies (ASYNCcnt) with an LDS
// double buffer: one s_wait_asynccnt + one barrier per 32-K step.
// ---------------------------------------------------------------------------

typedef __attribute__((ext_vector_type(16))) __bf16 v16bf;
typedef __attribute__((ext_vector_type(8)))  float  v8f;

union Frag16 {
  v16bf v;
  uint4 q[2];
  unsigned short u[16];
};

static __device__ __forceinline__ unsigned short f2bf(float f) {
  unsigned int u = __float_as_uint(f);
  unsigned int r = (u + 0x7FFFu + ((u >> 16) & 1u)) >> 16;  // RNE
  return (unsigned short)r;
}
static __device__ __forceinline__ float bf2f(unsigned short h) {
  return __uint_as_float(((unsigned int)h) << 16);
}

// 32-bit LDS offset of a __shared__ object (AS3 pointers are 32-bit).
static __device__ __forceinline__ unsigned lds_offset(const void* p) {
  return (unsigned)(unsigned long long)(__attribute__((address_space(3))) const char*)p;
}

// gfx1250 async copy: 16 bytes global -> LDS per lane, tracked by ASYNCcnt.
static __device__ __forceinline__ void async_ld_b128(unsigned ldsOfs,
                                                     unsigned long long gaddr) {
  asm volatile("global_load_async_to_lds_b128 %0, %1, off"
               :: "v"(ldsOfs), "v"(gaddr) : "memory");
}
static __device__ __forceinline__ void wait_async0() {
  asm volatile("s_wait_asynccnt 0" ::: "memory");
}

#define AT_ID 5
#define CITSEG_ID 7
#define BATCH 256
#define SEQ 512
#define HID 768

// ---------------------------------------------------------------------------
// Kernel 1: per-batch token scan -> meta[b] = {start, end, c_pos, has_c}
// ---------------------------------------------------------------------------
__global__ void meta_kernel(const int* __restrict__ tokens,
                            int* __restrict__ meta) {
  __shared__ int red[SEQ];
  const int b = blockIdx.x;
  const int t = threadIdx.x;           // 0..511
  const int BIG = SEQ + 1;
  const int tok = tokens[(size_t)b * SEQ + t];

  red[t] = (tok == AT_ID) ? t : BIG;
  __syncthreads();
  for (int o = SEQ / 2; o > 0; o >>= 1) {
    if (t < o) red[t] = min(red[t], red[t + o]);
    __syncthreads();
  }
  const int first = red[0];
  __syncthreads();

  red[t] = (tok == AT_ID && t > first) ? t : BIG;
  __syncthreads();
  for (int o = SEQ / 2; o > 0; o >>= 1) {
    if (t < o) red[t] = min(red[t], red[t + o]);
    __syncthreads();
  }
  const int second = red[0];
  __syncthreads();

  red[t] = (tok == CITSEG_ID) ? t : BIG;
  __syncthreads();
  for (int o = SEQ / 2; o > 0; o >>= 1) {
    if (t < o) red[t] = min(red[t], red[t + o]);
    __syncthreads();
  }
  const int cfirst = red[0];

  if (t == 0) {
    const bool ge2 = (second < BIG);
    meta[b * 4 + 0] = ge2 ? first : 0;
    meta[b * 4 + 1] = ge2 ? second : SEQ;
    meta[b * 4 + 2] = min(cfirst, SEQ - 1);
    meta[b * 4 + 3] = (cfirst < BIG) ? 1 : 0;
  }
}

// ---------------------------------------------------------------------------
// Kernel 2: masked max-pool + CITSEG gather (memory-bound floor: streams the
// 402 MB hidden_states once; start/end are block-uniform -> scalar branch).
// ---------------------------------------------------------------------------
__global__ void pool_gather_kernel(const float* __restrict__ hs,
                                   const int* __restrict__ meta,
                                   unsigned short* __restrict__ x,    // [256,1536]
                                   unsigned short* __restrict__ citH) // [256,768]
{
  const int b = blockIdx.x;
  const int t = threadIdx.x;           // 256 threads; 3 h-columns each
  const int start = meta[b * 4 + 0];
  const int end   = meta[b * 4 + 1];
  const int cpos  = meta[b * 4 + 2];
  const bool keepAny = (start > 0) || (end < SEQ - 1);

  const float* base = hs + (size_t)b * SEQ * HID;
  float m0 = -FLT_MAX, m1 = -FLT_MAX, m2 = -FLT_MAX;
  for (int s = 0; s < SEQ; ++s) {
    if (s < start || s > end) {
      const float* row = base + (size_t)s * HID;
      m0 = fmaxf(m0, row[t]);
      m1 = fmaxf(m1, row[t + 256]);
      m2 = fmaxf(m2, row[t + 512]);
    }
  }
  if (!keepAny) { m0 = 0.0f; m1 = 0.0f; m2 = 0.0f; }

  unsigned short* xr = x + (size_t)b * 1536;
  xr[t]       = f2bf(m0);
  xr[t + 256] = f2bf(m1);
  xr[t + 512] = f2bf(m2);

  const float* crow = base + (size_t)cpos * HID;
  unsigned short* cr = citH + (size_t)b * HID;
  cr[t]       = f2bf(crow[t]);
  cr[t + 256] = f2bf(crow[t + 256]);
  cr[t + 512] = f2bf(crow[t + 512]);
}

// ---------------------------------------------------------------------------
// Kernel 3: weight prep.  B[K,N] f32 -> Bt[Npad,Kpad] bf16 (transposed,
// zero-padded).  32x32 LDS-tiled transpose, coalesced both directions.
// ---------------------------------------------------------------------------
__global__ void convT_kernel(const float* __restrict__ B, int K, int N,
                             unsigned short* __restrict__ Bt, int Kpad) {
  __shared__ unsigned short tile[32][33];
  const int tx = threadIdx.x, ty = threadIdx.y;
  const int n0 = blockIdx.x * 32, k0 = blockIdx.y * 32;
#pragma unroll
  for (int i = 0; i < 4; ++i) {
    const int k = k0 + ty + i * 8;
    const int n = n0 + tx;
    const float v = (k < K && n < N) ? B[(size_t)k * N + n] : 0.0f;
    tile[ty + i * 8][tx] = f2bf(v);
  }
  __syncthreads();
#pragma unroll
  for (int i = 0; i < 4; ++i) {
    const int n = n0 + ty + i * 8;
    const int k = k0 + tx;
    Bt[(size_t)n * Kpad + k] = tile[tx][ty + i * 8];
  }
}

// ---------------------------------------------------------------------------
// Kernel 4: bf16 WMMA GEMM.  Out = act(A[256,Kpad] @ Bt^T + bias) as bf16.
// Block: 256 threads = 8 waves (wave32), block tile 128(M) x 64(N).
// Wave: 16 rows x 64 cols -> 4 accumulators, 4 WMMA per 32-K step.
// Double-buffered LDS filled by global_load_async_to_lds_b128 (ASYNCcnt):
// tile kt+1 streams into the spare buffer while tile kt feeds WMMA.
// mode: 0 = bias, 1 = bias+ReLU, 2 = bias then * has_c(row)
// ---------------------------------------------------------------------------
__global__ __launch_bounds__(256)
void gemm_bf16_kernel(const unsigned short* __restrict__ A, int lda,
                      const unsigned short* __restrict__ Bt, int ldb,
                      const float* __restrict__ bias, int Norig,
                      unsigned short* __restrict__ Out, int ldo, int outOfs,
                      int Ksteps, int mode, const int* __restrict__ meta)
{
  __shared__ __align__(16) unsigned short As[2][128 * 32]; // [buf][row][k]
  __shared__ __align__(16) unsigned short Bs[2][64 * 32];  // [buf][n][k]

  const int t    = threadIdx.x;
  const int wave = t >> 5;
  const int lane = t & 31;
  const int hsel = (lane < 16) ? 0 : 1;
  const int l16  = lane & 15;
  const int mBase = blockIdx.y * 128;
  const int nBase = blockIdx.x * 64;
  const int srow  = (t * 8) >> 5;      // 0..63 staging row
  const int scol  = (t * 8) & 31;      // {0,8,16,24}

  const unsigned short* aPtr0 = A  + (size_t)(mBase + srow)      * lda + scol;
  const unsigned short* aPtr1 = A  + (size_t)(mBase + srow + 64) * lda + scol;
  const unsigned short* bPtr  = Bt + (size_t)(nBase + srow)      * ldb + scol;

  // Per-lane LDS destinations (byte offsets) for the async copies.
  const unsigned aOfs0 = lds_offset(&As[0][srow * 32 + scol]);
  const unsigned aOfs1 = lds_offset(&As[0][(srow + 64) * 32 + scol]);
  const unsigned bOfs  = lds_offset(&Bs[0][srow * 32 + scol]);
  const unsigned aStride = (unsigned)(128 * 32 * sizeof(unsigned short));
  const unsigned bStride = (unsigned)(64 * 32 * sizeof(unsigned short));

  v8f acc[4];
  for (int i = 0; i < 4; ++i)
    for (int r = 0; r < 8; ++r) acc[i][r] = 0.0f;

  // Kick off tile 0 into buffer 0.
  async_ld_b128(aOfs0, (unsigned long long)aPtr0);
  async_ld_b128(aOfs1, (unsigned long long)aPtr1);
  async_ld_b128(bOfs,  (unsigned long long)bPtr);

  for (int kt = 0; kt < Ksteps; ++kt) {
    const int cur = kt & 1;
    wait_async0();                    // this wave's copies into 'cur' landed
    __syncthreads();                  // everyone's copies landed; prev reads done

    if (kt + 1 < Ksteps) {            // stream next tile into the spare buffer
      const int nxt = cur ^ 1;
      const int k0 = (kt + 1) * 32;
      async_ld_b128(aOfs0 + nxt * aStride, (unsigned long long)(aPtr0 + k0));
      async_ld_b128(aOfs1 + nxt * aStride, (unsigned long long)(aPtr1 + k0));
      async_ld_b128(bOfs  + nxt * bStride, (unsigned long long)(bPtr + k0));
    }

    // A fragment: lane<16 -> k {0..15}; lane>=16 -> k {16..31}.
    Frag16 a;
    const int arow = wave * 16 + l16;
    a.q[0] = *(const uint4*)(&As[cur][arow * 32 + hsel * 16]);
    a.q[1] = *(const uint4*)(&As[cur][arow * 32 + hsel * 16 + 8]);

    // Four B fragments in distinct registers so ds loads can issue together.
    const int bofs = l16 * 32 + hsel * 16;
    Frag16 bf0, bf1, bf2, bf3;
    bf0.q[0] = *(const uint4*)(&Bs[cur][bofs]);
    bf0.q[1] = *(const uint4*)(&Bs[cur][bofs + 8]);
    bf1.q[0] = *(const uint4*)(&Bs[cur][512 + bofs]);
    bf1.q[1] = *(const uint4*)(&Bs[cur][512 + bofs + 8]);
    bf2.q[0] = *(const uint4*)(&Bs[cur][1024 + bofs]);
    bf2.q[1] = *(const uint4*)(&Bs[cur][1024 + bofs + 8]);
    bf3.q[0] = *(const uint4*)(&Bs[cur][1536 + bofs]);
    bf3.q[1] = *(const uint4*)(&Bs[cur][1536 + bofs + 8]);

    acc[0] = __builtin_amdgcn_wmma_f32_16x16x32_bf16(
        false, a.v, false, bf0.v, (short)0, acc[0], false, false);
    acc[1] = __builtin_amdgcn_wmma_f32_16x16x32_bf16(
        false, a.v, false, bf1.v, (short)0, acc[1], false, false);
    acc[2] = __builtin_amdgcn_wmma_f32_16x16x32_bf16(
        false, a.v, false, bf2.v, (short)0, acc[2], false, false);
    acc[3] = __builtin_amdgcn_wmma_f32_16x16x32_bf16(
        false, a.v, false, bf3.v, (short)0, acc[3], false, false);
  }

  // Fused epilogue: bias / ReLU / has_c mask; pad columns written as zero.
#pragma unroll
  for (int nt = 0; nt < 4; ++nt) {
#pragma unroll
    for (int r = 0; r < 8; ++r) {
      const int mg = mBase + wave * 16 + r + hsel * 8;
      const int ng = nBase + nt * 16 + l16;
      float v = acc[nt][r];
      if (ng < Norig) v += bias[ng]; else v = 0.0f;
      if (mode == 1)      v = fmaxf(v, 0.0f);
      else if (mode == 2) v *= (meta[mg * 4 + 3] ? 1.0f : 0.0f);
      Out[(size_t)mg * ldo + outOfs + ng] = f2bf(v);
    }
  }
}

// ---------------------------------------------------------------------------
// Kernel 5: tiny head  out[256,6] = h2[256,3036] @ w3[3036,6] + b3   (f32)
// ---------------------------------------------------------------------------
__global__ void head_kernel(const unsigned short* __restrict__ h2,
                            const float* __restrict__ w3,
                            const float* __restrict__ b3,
                            float* __restrict__ out)
{
  __shared__ float red[256 * 6];
  const int b = blockIdx.x;
  const int t = threadIdx.x;
  float acc[6] = {0.f, 0.f, 0.f, 0.f, 0.f, 0.f};
  const unsigned short* hr = h2 + (size_t)b * 3072;
  for (int k = t; k < 3036; k += 256) {
    const float hv = bf2f(hr[k]);
#pragma unroll
    for (int n = 0; n < 6; ++n) acc[n] += hv * w3[(size_t)k * 6 + n];
  }
#pragma unroll
  for (int n = 0; n < 6; ++n) red[t * 6 + n] = acc[n];
  __syncthreads();
  for (int o = 128; o > 0; o >>= 1) {
    if (t < o) {
      for (int n = 0; n < 6; ++n) red[t * 6 + n] += red[(t + o) * 6 + n];
    }
    __syncthreads();
  }
  if (t < 6) out[b * 6 + t] = red[t] + b3[t];
}

// ---------------------------------------------------------------------------
extern "C" void kernel_launch(void* const* d_in, const int* in_sizes, int n_in,
                              void* d_out, int out_size, void* d_ws, size_t ws_size,
                              hipStream_t stream) {
  const int*   tokens = (const int*)  d_in[0];
  const float* hs     = (const float*)d_in[1];
  const float* proj_w = (const float*)d_in[2];
  const float* proj_b = (const float*)d_in[3];
  const float* enc_w  = (const float*)d_in[4];
  const float* enc_b  = (const float*)d_in[5];
  const float* w1     = (const float*)d_in[6];
  const float* b1     = (const float*)d_in[7];
  const float* w2     = (const float*)d_in[8];
  const float* b2     = (const float*)d_in[9];
  const float* w3     = (const float*)d_in[10];
  const float* b3     = (const float*)d_in[11];
  float* out = (float*)d_out;

  // Workspace layout (~34 MB): meta, bf16 activations, bf16 transposed weights.
  char* ws = (char*)d_ws;
  int* meta = (int*)ws;                                        // 256*4 ints
  unsigned short* citH  = (unsigned short*)(ws + 4096);        // [256, 768]
  unsigned short* cit   = citH  + (size_t)BATCH * 768;         // [256, 768]   (750 valid)
  unsigned short* xbuf  = cit   + (size_t)BATCH * 768;         // [256, 1536]  (1518 valid)
  unsigned short* h1b   = xbuf  + (size_t)BATCH * 1536;        // [256, 3072]  (3036 valid)
  unsigned short* h2b   = h1b   + (size_t)BATCH * 3072;        // [256, 3072]  (3036 valid)
  unsigned short* projT = h2b   + (size_t)BATCH * 3072;        // [768, 768]
  unsigned short* encT  = projT + (size_t)768 * 768;           // [768, 768]
  unsigned short* w1T   = encT  + (size_t)768 * 768;           // [3072, 1536]
  unsigned short* w2T   = w1T   + (size_t)3072 * 1536;         // [3072, 3072]

  // Weight prep (one-off ~85 MB of traffic => ~4 us at 23.3 TB/s).
  convT_kernel<<<dim3(24, 24), dim3(32, 8), 0, stream>>>(proj_w, 768, 750, projT, 768);
  convT_kernel<<<dim3(24, 24), dim3(32, 8), 0, stream>>>(enc_w, 750, 750, encT, 768);
  convT_kernel<<<dim3(96, 48), dim3(32, 8), 0, stream>>>(w1, 1518, 3036, w1T, 1536);
  convT_kernel<<<dim3(96, 96), dim3(32, 8), 0, stream>>>(w2, 3036, 3036, w2T, 3072);

  meta_kernel<<<BATCH, SEQ, 0, stream>>>(tokens, meta);
  pool_gather_kernel<<<BATCH, 256, 0, stream>>>(hs, meta, xbuf, citH);

  // cit = (citH @ proj_w + proj_b) * has_c      [Kpad=768, Npad=768, N=750]
  gemm_bf16_kernel<<<dim3(12, 2), 256, 0, stream>>>(
      citH, 768, projT, 768, proj_b, 750, cit, 768, 0, 24, 2, meta);

  // x[:, 768:1536] = cit @ enc_w + enc_b        [Kpad=768, Npad=768, N=750]
  gemm_bf16_kernel<<<dim3(12, 2), 256, 0, stream>>>(
      cit, 768, encT, 768, enc_b, 750, xbuf, 1536, 768, 24, 0, meta);

  // h1 = relu(x @ w1 + b1)                      [Kpad=1536, Npad=3072, N=3036]
  gemm_bf16_kernel<<<dim3(48, 2), 256, 0, stream>>>(
      xbuf, 1536, w1T, 1536, b1, 3036, h1b, 3072, 0, 48, 1, meta);

  // h2 = relu(h1 @ w2 + b2)                     [Kpad=3072, Npad=3072, N=3036]
  gemm_bf16_kernel<<<dim3(48, 2), 256, 0, stream>>>(
      h1b, 3072, w2T, 3072, b2, 3036, h2b, 3072, 0, 96, 1, meta);

  // out = h2 @ w3 + b3
  head_kernel<<<BATCH, 256, 0, stream>>>(h2b, w3, b3, out);
}
